// GNNModel_35914516529749
// MI455X (gfx1250) — matrix operand order repaired and checked by reading, not compiled, not run on hardware
//
#include <hip/hip_runtime.h>

typedef __attribute__((ext_vector_type(16))) _Float16 v16h;
typedef __attribute__((ext_vector_type(8)))  float    v8f;

#define HDIM 128

// ---------- WMMA helpers ----------
__device__ __forceinline__ v8f wmma_f16(v16h a, v16h b, v8f c) {
  // D = A(16x32 f16) * B(32x16 f16) + C(16x16 f32)
  return __builtin_amdgcn_wmma_f32_16x16x32_f16(
      false, a, false, b, (short)0, c, false, false);
}

// A-operand per-lane K mapping (matches packed B layout):
// lane h: slot i -> K = i + 16*h
__device__ __forceinline__ int kpermA(int i, int h) {
  return i + 16 * h;
}
// B-operand per-lane K mapping (contiguous k order, as packed by k0_pack_w/k0_pack_item):
// lane h: slot i -> K = i + 16*h   (lane&15 = column)

// ---------- K0a: pack weight matrix W[K][128] into WMMA B layout ----------
// dst[((kk*8 + ntile)*32 + lane)*16 + i] = (f16) W[(kk*32 + i + 16h)*128 + ntile*16 + (lane&15)]
__global__ void k0_pack_w(const float* __restrict__ W, _Float16* __restrict__ Wp,
                          int ksteps) {
  int idx = blockIdx.x * blockDim.x + threadIdx.x;
  int total = ksteps * 8 * 32;
  if (idx >= total) return;
  int lane = idx & 31;
  int nt   = (idx >> 5) & 7;
  int kk   = idx >> 8;
  int h = lane >> 4;
  int col = nt * 16 + (lane & 15);
  _Float16* dst = Wp + ((size_t)idx << 4);
#pragma unroll
  for (int i = 0; i < 16; ++i)
    dst[i] = (_Float16)W[(size_t)(kk * 32 + i + 16 * h) * HDIM + col];
}

// ---------- K0b: pack item_emb[n_item][128] into WMMA B layout (item idx = N) ----------
// dst[((nt*4 + kk)*32 + lane)*16 + i] = (f16) item_emb[(nt*16 + (lane&15))*128 + kk*32 + i + 16h]
__global__ void k0_pack_item(const float* __restrict__ item_emb,
                             _Float16* __restrict__ itemp, int n_item) {
  int idx = blockIdx.x * blockDim.x + threadIdx.x;
  int total = (n_item / 16) * 4 * 32;
  if (idx >= total) return;
  int lane = idx & 31;
  int kk   = (idx >> 5) & 3;
  int nt   = idx >> 7;
  int h = lane >> 4;
  int itrow = nt * 16 + (lane & 15);
  const float* src = item_emb + (size_t)itrow * HDIM + kk * 32 + 16 * h;
  _Float16* dst = itemp + ((size_t)idx << 4);
#pragma unroll
  for (int i = 0; i < 16; ++i) dst[i] = (_Float16)src[i];
}

// ---------- zero ----------
__global__ void k_zero4(float4* __restrict__ p, int n4) {
  int i = blockIdx.x * blockDim.x + threadIdx.x;
  if (i < n4) p[i] = make_float4(0.f, 0.f, 0.f, 0.f);
}

// ---------- K1: m = [item_emb[x], user_emb[userid[batch]]] @ W_g + b_g ----------
__global__ __launch_bounds__(256) void k1_gemm_m(
    const int* __restrict__ x, const int* __restrict__ userid,
    const float* __restrict__ item_emb, const float* __restrict__ user_emb,
    const _Float16* __restrict__ Wgp, const float* __restrict__ b_g,
    float* __restrict__ m_out, int ntiles, int npg) {
  int wave = (blockIdx.x * blockDim.x + threadIdx.x) >> 5;
  int lane = threadIdx.x & 31;
  if (wave >= ntiles) return;               // wave-uniform exit
  int row = lane & 15, h = lane >> 4;
  int node = wave * 16 + row;
  int g = node / npg;
  const float* hrow = item_emb + (size_t)x[node] * HDIM;
  const float* urow = user_emb + (size_t)userid[g] * HDIM;
  v16h A[8];
#pragma unroll
  for (int kk = 0; kk < 8; ++kk) {
#pragma unroll
    for (int i = 0; i < 16; ++i) {
      int kg = kk * 32 + kpermA(i, h);
      float v = (kg < HDIM) ? hrow[kg] : urow[kg - HDIM];
      A[kk][i] = (_Float16)v;
    }
  }
#pragma unroll 1
  for (int nt = 0; nt < 8; ++nt) {
    v8f c = {};
#pragma unroll
    for (int kk = 0; kk < 8; ++kk) {
      v16h b = *(const v16h*)(Wgp + (((size_t)(kk * 8 + nt) * 32 + lane) << 4));
      c = wmma_f16(A[kk], b, c);
    }
    int col = nt * 16 + row;
    float bg = b_g[col];
#pragma unroll
    for (int rr = 0; rr < 8; ++rr) {
      int mr = wave * 16 + rr + 8 * h;
      m_out[(size_t)mr * HDIM + col] = c[rr] + bg;
    }
  }
}

// ---------- K2: edge scatter: agg[dst]+=w_in*m[src]; agg[src]+=w_out*m[dst] ----------
__global__ void k2_edges(const int* __restrict__ edge_index,
                         const float* __restrict__ edge_count,
                         const float* __restrict__ in_deg,
                         const float* __restrict__ out_deg,
                         const float* __restrict__ m, float* __restrict__ agg,
                         int E) {
  int idx = blockIdx.x * blockDim.x + threadIdx.x;
  int e = idx >> 5;
  if (e >= E) return;
  int f = (idx & 31) << 2;
  int src = edge_index[e];
  int dst = edge_index[E + e];
  float ec = edge_count[e];
  float wi = ec * in_deg[e];
  float wo = ec * out_deg[e];
  const float4 ms = *(const float4*)(m + (size_t)src * HDIM + f);
  const float4 md = *(const float4*)(m + (size_t)dst * HDIM + f);
  float* ad = agg + (size_t)dst * HDIM + f;
  float* as = agg + (size_t)src * HDIM + f;
  unsafeAtomicAdd(ad + 0, wi * ms.x);
  unsafeAtomicAdd(ad + 1, wi * ms.y);
  unsafeAtomicAdd(ad + 2, wi * ms.z);
  unsafeAtomicAdd(ad + 3, wi * ms.w);
  unsafeAtomicAdd(as + 0, wo * md.x);
  unsafeAtomicAdd(as + 1, wo * md.y);
  unsafeAtomicAdd(as + 2, wo * md.z);
  unsafeAtomicAdd(as + 3, wo * md.w);
}

// ---------- K3: hidden = tanh(agg) ----------
__global__ void k3_tanh(const float4* __restrict__ agg, float4* __restrict__ hidden,
                        int n4) {
  int i = blockIdx.x * blockDim.x + threadIdx.x;
  if (i >= n4) return;
  float4 a = agg[i];
  float4 r;
  r.x = tanhf(a.x); r.y = tanhf(a.y); r.z = tanhf(a.z); r.w = tanhf(a.w);
  hidden[i] = r;
}

// ---------- K4: per-graph attention pooling + s_h epilogue ----------
// one block (128 thr = 4 waves) per graph; npg <= 64
__global__ __launch_bounds__(128) void k4_graph(
    const float* __restrict__ hidden, const int* __restrict__ userid,
    const float* __restrict__ user_emb, const _Float16* __restrict__ W2p,
    const float* __restrict__ b2, const float* __restrict__ W1,
    const float* __restrict__ b1, const float* __restrict__ num_count,
    const float* __restrict__ W5, const float* __restrict__ b5,
    const float* __restrict__ Wu, const float* __restrict__ bu,
    _Float16* __restrict__ s_h16, int npg, int N) {
  __shared__ float hl[HDIM];          // hidden[last node]
  __shared__ float us[HDIM];          // user vector
  __shared__ float inner[64][HDIM];   // sigmoid(X@W2+b2)
  __shared__ float coef[64];          // num_count * alpha (0 padded)
  __shared__ float sg[HDIM];          // pooled s_g
  int g = blockIdx.x;
  int tid = threadIdx.x;
  int base = g * npg;
  hl[tid] = hidden[(size_t)(base + npg - 1) * HDIM + tid];
  us[tid] = user_emb[(size_t)userid[g] * HDIM + tid];
  __syncthreads();

  // Phase A: inner = sigmoid([v_n, hidden, u] @ W2 + b2) via WMMA (K=384)
  int wave = tid >> 5, lane = tid & 31;
  int row = lane & 15, h = lane >> 4;
  int r = wave * 16 + row;
  int node = base + r;
  if (node >= N) node = N - 1;        // clamp; rows >= npg masked later
  const float* hn = hidden + (size_t)node * HDIM;
  v16h A[12];
#pragma unroll
  for (int kk = 0; kk < 12; ++kk) {
#pragma unroll
    for (int i = 0; i < 16; ++i) {
      int kg = kk * 32 + kpermA(i, h);
      float v = (kg < HDIM) ? hl[kg]
              : (kg < 2 * HDIM) ? hn[kg - HDIM]
                                : us[kg - 2 * HDIM];
      A[kk][i] = (_Float16)v;
    }
  }
#pragma unroll 1
  for (int nt = 0; nt < 8; ++nt) {
    v8f c = {};
#pragma unroll
    for (int kk = 0; kk < 12; ++kk) {
      v16h b = *(const v16h*)(W2p + (((size_t)(kk * 8 + nt) * 32 + lane) << 4));
      c = wmma_f16(A[kk], b, c);
    }
    int col = nt * 16 + row;
    float bb = b2[col];
#pragma unroll
    for (int rr = 0; rr < 8; ++rr) {
      int mr = wave * 16 + rr + 8 * h;
      float v = c[rr] + bb;
      inner[mr][col] = 1.f / (1.f + __expf(-v));
    }
  }
  __syncthreads();

  // Phase B: alpha + coef
  if (tid < 64) {
    float a = b1[0];
#pragma unroll 4
    for (int cidx = 0; cidx < HDIM; ++cidx) a += inner[tid][cidx] * W1[cidx];
    coef[tid] = (tid < npg) ? num_count[base + tid] * a : 0.f;
  }
  __syncthreads();

  // Phase C: s_g[col] = sum_r coef[r] * hidden[base+r][col]
  {
    float s = 0.f;
    for (int rr = 0; rr < npg; ++rr)
      s += coef[rr] * hidden[(size_t)(base + rr) * HDIM + tid];
    sg[tid] = s;
  }
  __syncthreads();

  // Phase D: s_h = [h_last, s_g]@W5 + b5 + tanh(u@Wu + bu)
  {
    int col = tid;
    float v = b5[col];
#pragma unroll 4
    for (int k = 0; k < HDIM; ++k) v += hl[k] * W5[(size_t)k * HDIM + col];
#pragma unroll 4
    for (int k = 0; k < HDIM; ++k) v += sg[k] * W5[(size_t)(HDIM + k) * HDIM + col];
    float uu = bu[col];
#pragma unroll 4
    for (int k = 0; k < HDIM; ++k) uu += us[k] * Wu[(size_t)k * HDIM + col];
    v += tanhf(uu);
    s_h16[(size_t)g * HDIM + col] = (_Float16)v;
  }
}

// ---------- K5: out = s_h @ item_emb.T  (M=B, N=n_item, K=128) ----------
// Output (205 MB) is write-once streaming -> non-temporal stores so it does
// not evict the L2-resident 12.8 MB packed item_emb operand.
__global__ __launch_bounds__(256) void k5_out(const _Float16* __restrict__ s_h16,
                                              const _Float16* __restrict__ itemp,
                                              float* __restrict__ out, int n_item) {
  int mtile = blockIdx.y;
  int lane = threadIdx.x & 31;
  int wavein = threadIdx.x >> 5;
  int row = lane & 15, h = lane >> 4;
  const _Float16* arow = s_h16 + (size_t)(mtile * 16 + row) * HDIM;
  v16h A[4];
#pragma unroll
  for (int kk = 0; kk < 4; ++kk)
#pragma unroll
    for (int i = 0; i < 16; ++i)
      A[kk][i] = arow[kk * 32 + kpermA(i, h)];
  int ntiles = n_item / 16;
  for (int nt = blockIdx.x * 8 + wavein; nt < ntiles; nt += gridDim.x * 8) {
    v8f c = {};
#pragma unroll
    for (int kk = 0; kk < 4; ++kk) {
      v16h b = *(const v16h*)(itemp + (((size_t)nt * 4 + kk) * 32 + lane) * 16);
      c = wmma_f16(A[kk], b, c);
    }
    int col = nt * 16 + row;
#pragma unroll
    for (int rr = 0; rr < 8; ++rr) {
      int mr = mtile * 16 + rr + 8 * h;
      __builtin_nontemporal_store(c[rr], &out[(size_t)mr * n_item + col]);
    }
  }
}

// ---------- launch ----------
extern "C" void kernel_launch(void* const* d_in, const int* in_sizes, int n_in,
                              void* d_out, int out_size, void* d_ws, size_t ws_size,
                              hipStream_t stream) {
  const int*   x          = (const int*)d_in[0];
  const int*   edge_index = (const int*)d_in[1];
  const float* edge_count = (const float*)d_in[3];
  const float* in_deg     = (const float*)d_in[4];
  const float* out_deg    = (const float*)d_in[5];
  const float* num_count  = (const float*)d_in[6];
  const int*   userid     = (const int*)d_in[7];
  const float* item_emb   = (const float*)d_in[9];
  const float* user_emb   = (const float*)d_in[10];
  const float* W_g = (const float*)d_in[11];
  const float* b_g = (const float*)d_in[12];
  const float* W1  = (const float*)d_in[13];
  const float* b1  = (const float*)d_in[14];
  const float* W2  = (const float*)d_in[15];
  const float* b2  = (const float*)d_in[16];
  const float* W5  = (const float*)d_in[17];
  const float* b5  = (const float*)d_in[18];
  const float* Wu  = (const float*)d_in[19];
  const float* bu  = (const float*)d_in[20];
  float* out = (float*)d_out;

  const int N      = in_sizes[0];            // 51200
  const int E      = in_sizes[3];            // 102400
  const int B      = in_sizes[7];            // 1024
  const int n_item = in_sizes[9] / HDIM;     // 50000
  const int npg    = N / B;                  // 50

  // workspace carve (256B aligned)
  char* ws = (char*)d_ws;
  size_t off = 0;
  auto carve = [&](size_t bytes) -> char* {
    char* p = ws + off;
    off = (off + bytes + 255) & ~(size_t)255;
    return p;
  };
  _Float16* itemp = (_Float16*)carve((size_t)n_item * HDIM * sizeof(_Float16));
  _Float16* Wgp   = (_Float16*)carve((size_t)8 * 8 * 32 * 16 * sizeof(_Float16));
  _Float16* W2p   = (_Float16*)carve((size_t)12 * 8 * 32 * 16 * sizeof(_Float16));
  float*    mbuf  = (float*)carve((size_t)N * HDIM * sizeof(float));
  float*    agg   = (float*)carve((size_t)N * HDIM * sizeof(float));
  _Float16* s_h16 = (_Float16*)carve((size_t)B * HDIM * sizeof(_Float16));
  float*    hidden = mbuf;                   // m dead after K2; reuse buffer

  const int T = 256;
  // K0: pack operands
  int itemWork = (n_item / 16) * 4 * 32;
  k0_pack_item<<<(itemWork + T - 1) / T, T, 0, stream>>>(item_emb, itemp, n_item);
  k0_pack_w<<<(8 * 8 * 32 + T - 1) / T, T, 0, stream>>>(W_g, Wgp, 8);
  k0_pack_w<<<(12 * 8 * 32 + T - 1) / T, T, 0, stream>>>(W2, W2p, 12);
  // zero agg
  int n4 = N * HDIM / 4;
  k_zero4<<<(n4 + T - 1) / T, T, 0, stream>>>((float4*)agg, n4);
  // K1: m GEMM
  int ntiles = N / 16;
  k1_gemm_m<<<(ntiles + 7) / 8, T, 0, stream>>>(x, userid, item_emb, user_emb,
                                                Wgp, b_g, mbuf, ntiles, npg);
  // K2: edge scatter
  long long ethreads = (long long)E * 32;
  k2_edges<<<(int)((ethreads + T - 1) / T), T, 0, stream>>>(
      edge_index, edge_count, in_deg, out_deg, mbuf, agg, E);
  // K3: tanh (agg -> hidden, aliasing m buffer)
  k3_tanh<<<(n4 + T - 1) / T, T, 0, stream>>>((const float4*)agg, (float4*)hidden, n4);
  // K4: per-graph attention pooling + s_h
  k4_graph<<<B, 128, 0, stream>>>(hidden, userid, user_emb, W2p, b2, W1, b1,
                                  num_count, W5, b5, Wu, bu, s_h16, npg, N);
  // K5: final big GEMM
  k5_out<<<dim3(64, B / 16), T, 0, stream>>>(s_h16, itemp, out, n_item);
}
